// predMainLSTMPT_893353198023
// MI455X (gfx1250) — matrix operand order repaired and checked by reading, not compiled, not run on hardware
//
#include <hip/hip_runtime.h>
#include <hip/hip_bf16.h>
#include <math.h>

// ---------------------------------------------------------------------------
// predMainLSTMPT on MI455X (gfx1250): 2-layer LSTM, B=8192 T=50 F=25 H1=200 H2=100
// Layer-1 recurrent GEMM done with v_wmma_f32_16x16x32_bf16, batch-tiled so the
// whole time recurrence runs inside one workgroup (no global sync).
// 8 waves/WG + ping-pong B-fragment buffers for load/WMMA overlap.
// ---------------------------------------------------------------------------

typedef __attribute__((ext_vector_type(16))) __bf16 v16bf;
typedef __attribute__((ext_vector_type(8)))  float  v8f;

#define WGS 256          // 8 waves (wave32)
#define MB  32           // batch rows per workgroup
#define LDA 320          // A-operand LDS row stride (halves); covers K1P=256 and K2P=320

// Problem constants
#define TT   50
#define FF   25
#define H1   200
#define H2   100
#define G1   800         // 4*H1
#define G2   400         // 4*H2
#define NT1  50          // G1/16
#define NT2  25          // G2/16
#define EPT  25          // elementwise cell elems per thread: 32*200/256

__device__ __forceinline__ float fast_tanh(float x) {
#if __has_builtin(__builtin_amdgcn_tanhf)
  return __builtin_amdgcn_tanhf(x);          // V_TANH_F32 (gfx1250 TRANS op)
#else
  return tanhf(x);
#endif
}

__device__ __forceinline__ float fast_rcp(float x) {
#if __has_builtin(__builtin_amdgcn_rcpf)
  return __builtin_amdgcn_rcpf(x);           // V_RCP_F32
#else
  return 1.0f / x;
#endif
}

__device__ __forceinline__ float sigf(float v) {
  return fast_rcp(1.0f + __expf(-v));
}

// A-matrix 16x32 bf16 fragment (contiguous K order, matches B packing):
// lane L: M = L&15; kb = (L>>4)*16; VGPR vg in 0..7 -> K = kb+2vg(+1).
// Loaded as 8x b32 from LDS.
__device__ __forceinline__ v16bf load_a_frag(const __bf16* As, int mrow0,
                                             int ktile, int lane) {
  int m  = mrow0 + (lane & 15);
  int kb = (lane >> 4) * 16;
  union { v16bf v; unsigned int u[8]; } r;
  const __bf16* row = As + m * LDA;
#pragma unroll
  for (int vg = 0; vg < 8; ++vg) {
    int k = ktile * 32 + kb + 2 * vg;
    r.u[vg] = *(const unsigned int*)(row + k);   // k even -> 4B aligned
  }
  return r.v;
}

// B fragments are pre-packed fragment-major: one coalesced 32B load per lane.
__device__ __forceinline__ v16bf load_b_frag(const __bf16* wf, int kt, int nt,
                                             int NT, int lane) {
  return *(const v16bf*)(wf + ((((kt * NT + nt) << 5) + lane) << 4));
}

// ---------------------------------------------------------------------------
// Weight pre-pack kernels: Wt[k][n] = Wcomb[n][k] in B-matrix (32x16 bf16)
// fragment-major layout.  B layout used here (contiguous K): lane L: N = L&15, K-half =
// (L>>4)*16, element j in 0..15 -> K = khalf + j.
// ---------------------------------------------------------------------------
__global__ void pack_w1(const float* __restrict__ W_ih1,
                        const float* __restrict__ W_hh1,
                        __bf16* __restrict__ wf) {
  int e = blockIdx.x * blockDim.x + threadIdx.x;
  if (e >= 8 * NT1 * 512) return;              // 204800 halves
  int kt   = e / (NT1 * 512);
  int r    = e % (NT1 * 512);
  int nt   = r / 512;
  int r2   = r % 512;
  int lane = r2 >> 4;
  int j    = r2 & 15;
  int n = nt * 16 + (lane & 15);
  int k = kt * 32 + ((lane >> 4) << 4) + j;
  float v = 0.0f;
  if (k < FF)            v = W_ih1[n * FF + k];
  else if (k < FF + H1)  v = W_hh1[n * H1 + (k - FF)];
  wf[e] = (__bf16)v;
}

__global__ void pack_w2(const float* __restrict__ W_ih2,
                        const float* __restrict__ W_hh2,
                        __bf16* __restrict__ wf) {
  int e = blockIdx.x * blockDim.x + threadIdx.x;
  if (e >= 10 * NT2 * 512) return;             // 128000 halves
  int kt   = e / (NT2 * 512);
  int r    = e % (NT2 * 512);
  int nt   = r / 512;
  int r2   = r % 512;
  int lane = r2 >> 4;
  int j    = r2 & 15;
  int n = nt * 16 + (lane & 15);
  int k = kt * 32 + ((lane >> 4) << 4) + j;
  float v = 0.0f;
  if (k < H1)            v = W_ih2[n * H1 + k];
  else if (k < H1 + H2)  v = W_hh2[n * H2 + (k - H1)];
  wf[e] = (__bf16)v;
}

// ---------------------------------------------------------------------------
// Main kernel: one WG = 32 batch rows, full 50-step recurrence + layer2 + fc.
// Dynamic LDS layout (127,680 B total):
//   [0      .. 20480)  A1    : 32 x 320 bf16 (x_t cols 0..24, h cols 25..224)
//   [20480  ..122880)  gbuf  : layer1 gates 32x800 f32; reused for layer2:
//                              gates2 f32 @0..51200, h2buf @51200..64000,
//                              A2 bf16 @65536..86016
//   [122880..126080)  bias1 : 800 f32  (b_ih1+b_hh1)
//   [126080..127680)  bias2 : 400 f32  (b_ih2+b_hh2)
// Biases are folded into the WMMA accumulator init (broadcast over M rows).
// ---------------------------------------------------------------------------
__global__ __launch_bounds__(WGS) void lstm_main(
    const float* __restrict__ x,
    const float* __restrict__ b_ih1, const float* __restrict__ b_hh1,
    const float* __restrict__ b_ih2, const float* __restrict__ b_hh2,
    const float* __restrict__ fc_w,  const float* __restrict__ fc_b,
    const float* __restrict__ h1_0,  const float* __restrict__ c1_0,
    const float* __restrict__ h2_0,  const float* __restrict__ c2_0,
    const __bf16* __restrict__ wfrag1, const __bf16* __restrict__ wfrag2,
    float* __restrict__ out) {
  extern __shared__ char smem[];
  __bf16* A1     = (__bf16*)smem;
  char*   gbuf   = smem + 20480;
  float*  gates  = (float*)gbuf;
  float*  bias1  = (float*)(smem + 122880);
  float*  bias2  = (float*)(smem + 126080);
  float*  gates2 = (float*)gbuf;
  float*  h2buf  = (float*)(gbuf + 51200);
  __bf16* A2     = (__bf16*)(gbuf + 65536);

  const int tid    = threadIdx.x;
  const int lane   = tid & 31;
  const int w      = tid >> 5;        // wave 0..7
  const int wgbase = blockIdx.x * MB;

  // ---- init: biases, zero A1 (incl. K padding), h1_0, c1_0 (registers) ----
  for (int i = tid; i < G1; i += WGS) bias1[i] = b_ih1[i] + b_hh1[i];
  for (int i = tid; i < G2; i += WGS) bias2[i] = b_ih2[i] + b_hh2[i];
  for (int i = tid; i < MB * LDA; i += WGS) A1[i] = (__bf16)0.0f;
  __syncthreads();
  for (int i = tid; i < MB * H1; i += WGS) {
    int b = i / H1, j = i % H1;
    A1[b * LDA + FF + j] = (__bf16)h1_0[(wgbase + b) * H1 + j];
  }
  float creg[EPT];                     // c-state: 32*200 / 256 = 25 per thread
#pragma unroll
  for (int r = 0; r < EPT; ++r) {
    int idx = tid + WGS * r;
    creg[r] = c1_0[(wgbase + idx / H1) * H1 + (idx % H1)];
  }

  // wave tiling: 2 m-tiles x 50 n-tiles over 8 waves; mt fixed per wave so
  // A fragments are loaded once per timestep, n-tiles strided by 4.
  const int mt     = w & 1;
  const int nt0    = w >> 1;                       // 0..3
  const int ntiles = (NT1 - nt0 + 3) / 4;          // 12 or 13 (wave-uniform)

  // =================== layer-1 time recurrence ===================
  for (int t = 0; t < TT; ++t) {
    // stage x_t into A1 cols 0..24 (bf16)
    for (int i = tid; i < MB * FF; i += WGS) {
      int b = i / FF, f = i % FF;
      A1[b * LDA + f] = (__bf16)x[(wgbase + b) * (TT * FF) + t * FF + f];
    }
    __syncthreads();   // x staged AND prev-step h written before GEMM

    // gates[32,800] = A1[32,256] x Wt1[256,800] + bias1  (bf16 WMMA, f32 acc)
    v16bf afr[8];
#pragma unroll
    for (int kt = 0; kt < 8; ++kt) afr[kt] = load_a_frag(A1, mt * 16, kt, lane);

    auto loadB = [&](v16bf (&bf)[8], int nt) {
#pragma unroll
      for (int kt = 0; kt < 8; ++kt) bf[kt] = load_b_frag(wfrag1, kt, nt, NT1, lane);
    };
    auto gemm_tile = [&](const v16bf (&bf)[8], int nt) {
      int n = nt * 16 + (lane & 15);
      float bv = bias1[n];                         // bias folded into C init
      v8f acc = {bv, bv, bv, bv, bv, bv, bv, bv};
#pragma unroll
      for (int kt = 0; kt < 8; ++kt)
        acc = __builtin_amdgcn_wmma_f32_16x16x32_bf16(
            false, afr[kt], false, bf[kt], (short)0, acc, false, false);
      // C/D layout: lane: N = lane&15; VGPR r -> M = r + (lane>=16 ? 8 : 0)
      int mrow = mt * 16 + ((lane >> 4) << 3);
#pragma unroll
      for (int r8 = 0; r8 < 8; ++r8) gates[(mrow + r8) * G1 + n] = acc[r8];
    };

    // Ping-pong pipeline: loads for tile i+1 issue before the WMMA chain of
    // tile i; overlapping buffer lifetimes keep them in distinct registers.
    v16bf b0[8], b1[8];
    loadB(b0, nt0);
    int i = 0;
    for (; i + 1 < ntiles; i += 2) {
      loadB(b1, nt0 + 4 * (i + 1));
      gemm_tile(b0, nt0 + 4 * i);
      if (i + 2 < ntiles) loadB(b0, nt0 + 4 * (i + 2));
      gemm_tile(b1, nt0 + 4 * (i + 1));
    }
    if (i < ntiles) gemm_tile(b0, nt0 + 4 * i);
    __syncthreads();

    // elementwise LSTM cell; new h (bf16) written straight into A1
#pragma unroll 1
    for (int r = 0; r < EPT; ++r) {
      int idx = tid + WGS * r;
      int b = idx / H1, hh = idx % H1;
      const float* g = gates + b * G1;
      float iv = sigf(g[hh]);
      float fv = sigf(g[H1 + hh]);
      float gv = fast_tanh(g[2 * H1 + hh]);
      float ov = sigf(g[3 * H1 + hh]);
      float c  = fv * creg[r] + iv * gv;
      creg[r]  = c;
      A1[b * LDA + FF + hh] = (__bf16)(ov * fast_tanh(c));
    }
    __syncthreads();
  }

  // =================== layer 2 (single step, stale hidden2) ===================
  for (int i = tid; i < MB * LDA; i += WGS) A2[i] = (__bf16)0.0f;
  __syncthreads();
  for (int i = tid; i < MB * H1; i += WGS) {
    int b = i / H1, j = i % H1;
    A2[b * LDA + j] = A1[b * LDA + FF + j];          // h_last
  }
  for (int i = tid; i < MB * H2; i += WGS) {
    int b = i / H2, j = i % H2;
    A2[b * LDA + H1 + j] = (__bf16)h2_0[(wgbase + b) * H2 + j];
  }
  __syncthreads();

  // gates2[32,400] = A2[32,320] x Wt2[320,400] + bias2; 50 tiles over 8 waves
  for (int tt = w; tt < 2 * NT2; tt += 8) {
    int mt2 = tt / NT2, nt2 = tt % NT2;
    int n   = nt2 * 16 + (lane & 15);
    float bv = bias2[n];
    v8f acc = {bv, bv, bv, bv, bv, bv, bv, bv};
#pragma unroll
    for (int kt = 0; kt < 10; ++kt) {
      v16bf a2 = load_a_frag(A2, mt2 * 16, kt, lane);
      v16bf b2 = load_b_frag(wfrag2, kt, nt2, NT2, lane);
      acc = __builtin_amdgcn_wmma_f32_16x16x32_bf16(
          false, a2, false, b2, (short)0, acc, false, false);
    }
    int mrow = mt2 * 16 + ((lane >> 4) << 3);
#pragma unroll
    for (int r8 = 0; r8 < 8; ++r8) gates2[(mrow + r8) * G2 + n] = acc[r8];
  }
  __syncthreads();

  for (int i = tid; i < MB * H2; i += WGS) {
    int b = i / H2, hh = i % H2;
    const float* g = gates2 + b * G2;
    float iv = sigf(g[hh]);
    float fv = sigf(g[H2 + hh]);
    float gv = fast_tanh(g[2 * H2 + hh]);
    float ov = sigf(g[3 * H2 + hh]);
    float c  = fv * c2_0[(wgbase + b) * H2 + hh] + iv * gv;
    h2buf[b * H2 + hh] = ov * fast_tanh(c);
  }
  __syncthreads();

  // final fc + sigmoid: one thread per batch row
  if (tid < MB) {
    float s = fc_b[0];
    for (int k = 0; k < H2; ++k) s += h2buf[tid * H2 + k] * fc_w[k];
    out[wgbase + tid] = sigf(s);
  }
}

// ---------------------------------------------------------------------------
extern "C" void kernel_launch(void* const* d_in, const int* in_sizes, int n_in,
                              void* d_out, int out_size, void* d_ws,
                              size_t ws_size, hipStream_t stream) {
  (void)in_sizes; (void)n_in; (void)out_size; (void)ws_size;
  const float* x     = (const float*)d_in[0];
  const float* W_ih1 = (const float*)d_in[1];
  const float* W_hh1 = (const float*)d_in[2];
  const float* b_ih1 = (const float*)d_in[3];
  const float* b_hh1 = (const float*)d_in[4];
  const float* W_ih2 = (const float*)d_in[5];
  const float* W_hh2 = (const float*)d_in[6];
  const float* b_ih2 = (const float*)d_in[7];
  const float* b_hh2 = (const float*)d_in[8];
  const float* fc_w  = (const float*)d_in[9];
  const float* fc_b  = (const float*)d_in[10];
  const float* h1_0  = (const float*)d_in[11];
  const float* c1_0  = (const float*)d_in[12];
  const float* h2_0  = (const float*)d_in[13];
  const float* c2_0  = (const float*)d_in[14];
  float* out = (float*)d_out;

  __bf16* wf1 = (__bf16*)d_ws;                          // 409,600 B
  __bf16* wf2 = (__bf16*)((char*)d_ws + 409600);        // 256,000 B

  pack_w1<<<(8 * NT1 * 512 + 255) / 256, 256, 0, stream>>>(W_ih1, W_hh1, wf1);
  pack_w2<<<(10 * NT2 * 512 + 255) / 256, 256, 0, stream>>>(W_ih2, W_hh2, wf2);

  const size_t smem_bytes = 127680;
  lstm_main<<<8192 / MB, WGS, smem_bytes, stream>>>(
      x, b_ih1, b_hh1, b_ih2, b_hh2, fc_w, fc_b,
      h1_0, c1_0, h2_0, c2_0, wf1, wf2, out);
}
